// GPT2ParallelSelfAttention_50379966382622
// MI455X (gfx1250) — matrix operand order repaired and checked by reading, not compiled, not run on hardware
//
#include <hip/hip_runtime.h>
#include <hip/hip_bf16.h>
#include <math.h>

// ---------------------------------------------------------------------------
// CDNA5 (gfx1250) sparse 2D attention.
// GEMMs on v_wmma_f32_16x16x32_bf16 (wave32), software-pipelined global->LDS
// staging with packed bf16 conversion, A/B fragments in
// contiguous k order (k = 16*(lane/16) + s). Window ops on vectorized VALU.
// ---------------------------------------------------------------------------

typedef __attribute__((ext_vector_type(16))) __bf16 bf16x16;
typedef __attribute__((ext_vector_type(8)))  __bf16 bf16x8;
typedef __attribute__((ext_vector_type(4)))  __bf16 bf16x4;
typedef __attribute__((ext_vector_type(2)))  __bf16 bf16x2;
typedef __attribute__((ext_vector_type(8)))  float  f32x8;

#define TM 128
#define TK 32
#define LDS_K 40   // 32 + 8 pad; 80B row pitch keeps b128/b64 frag reads aligned

// ---------------------------------------------------------------------------
// Pipelined WMMA tile core. 256 threads = 8 waves.
//   WGN = waves along N. TN = WGN*32. Waves along M = 8/WGN.
//   Per-wave: MI (=WGN) m-subtiles x 2 n-subtiles of 16x16 -> MI*2 WMMAs/K-step.
//   bKN==1: B global row-major [K][N] (weights) -> transposed into LDS [N][K]
//           with packed bf16x2 stores (pairs along K).
//   bKN==0: B global row-major [N][K] -> straight bf16x4 copy.
// A rows beyond Mact are address-clamped (their D rows are epilogue-guarded).
// Pipeline: stage(regs->LDS); barrier; fetch(next)->regs; wmma; barrier.
// Inner compute keeps only one A-fragment live (B frags hoisted) to stay
// within the VGPR budget without spilling accumulators.
// ---------------------------------------------------------------------------
template<int WGN>
__device__ __forceinline__ void wmma_pipe(
    const float* __restrict__ A, int lda, int Mact, int mBase,
    const float* __restrict__ B, int ldb, int bKN, int nBase,
    int K, __bf16* As, __bf16* Bs, f32x8 (&acc)[WGN][2])
{
  constexpr int TNv    = WGN * 32;
  constexpr int MI     = WGN;             // m-subtiles per wave
  constexpr int BKN_IT = (4 * TNv) / 256; // 2x4 blocks per thread-iter
  constexpr int BNK_IT = (TNv * 8) / 256; // float4 groups per thread-iter
  constexpr int BR     = 2 * BKN_IT;      // == BNK_IT for both WGN=2,4

  const int tid  = threadIdx.x;
  const int lane = tid & 31;
  const int wid  = tid >> 5;
  const int wm   = wid / WGN;
  const int wn   = wid % WGN;

#pragma unroll
  for (int i = 0; i < MI; ++i)
#pragma unroll
    for (int j = 0; j < 2; ++j)
#pragma unroll
      for (int q = 0; q < 8; ++q) acc[i][j][q] = 0.0f;

  float4 aR[4];
  float4 bR[BR];

  auto fetch = [&](int k0) {
#pragma unroll
    for (int v = 0; v < 4; ++v) {
      int idx = tid + v * 256;            // float4 group, 0..1023
      int row = idx >> 3;
      int kg  = (idx & 7) << 2;
      int gr  = mBase + row;
      if (gr >= Mact) gr = Mact - 1;      // clamp; guarded rows never stored
      aR[v] = *(const float4*)(A + (size_t)gr * lda + k0 + kg);
    }
    if (bKN) {
#pragma unroll
      for (int v = 0; v < BKN_IT; ++v) {
        int idx = tid + v * 256;
        int kp  = idx / (TNv / 4);        // 0..15 -> K pair base 2*kp
        int ng  = (idx % (TNv / 4)) << 2;
        const float* p = B + (size_t)(k0 + 2 * kp) * ldb + nBase + ng;
        bR[2 * v]     = *(const float4*)p;
        bR[2 * v + 1] = *(const float4*)(p + ldb);
      }
    } else {
#pragma unroll
      for (int v = 0; v < BNK_IT; ++v) {
        int idx = tid + v * 256;
        int nr  = idx >> 3;
        int kg  = (idx & 7) << 2;
        bR[v] = *(const float4*)(B + (size_t)(nBase + nr) * ldb + k0 + kg);
      }
    }
  };

  auto stage = [&]() {
#pragma unroll
    for (int v = 0; v < 4; ++v) {
      int idx = tid + v * 256;
      int row = idx >> 3;
      int kg  = (idx & 7) << 2;
      bf16x4 d = { (__bf16)aR[v].x, (__bf16)aR[v].y,
                   (__bf16)aR[v].z, (__bf16)aR[v].w };
      *(bf16x4*)(As + row * LDS_K + kg) = d;
    }
    if (bKN) {
#pragma unroll
      for (int v = 0; v < BKN_IT; ++v) {
        int idx = tid + v * 256;
        int kp  = idx / (TNv / 4);
        int ng  = (idx % (TNv / 4)) << 2;
        int k   = 2 * kp;
        float4 f0 = bR[2 * v], f1 = bR[2 * v + 1];
        bf16x2 p0 = { (__bf16)f0.x, (__bf16)f1.x };
        bf16x2 p1 = { (__bf16)f0.y, (__bf16)f1.y };
        bf16x2 p2 = { (__bf16)f0.z, (__bf16)f1.z };
        bf16x2 p3 = { (__bf16)f0.w, (__bf16)f1.w };
        *(bf16x2*)(Bs + (ng + 0) * LDS_K + k) = p0;
        *(bf16x2*)(Bs + (ng + 1) * LDS_K + k) = p1;
        *(bf16x2*)(Bs + (ng + 2) * LDS_K + k) = p2;
        *(bf16x2*)(Bs + (ng + 3) * LDS_K + k) = p3;
      }
    } else {
#pragma unroll
      for (int v = 0; v < BNK_IT; ++v) {
        int idx = tid + v * 256;
        int nr  = idx >> 3;
        int kg  = (idx & 7) << 2;
        bf16x4 d = { (__bf16)bR[v].x, (__bf16)bR[v].y,
                     (__bf16)bR[v].z, (__bf16)bR[v].w };
        *(bf16x4*)(Bs + nr * LDS_K + kg) = d;
      }
    }
  };

  fetch(0);
  for (int k0 = 0; k0 < K; k0 += TK) {
    stage();
    __syncthreads();
    if (k0 + TK < K) fetch(k0 + TK);     // overlap next fetch with WMMAs

    // Fragments (wave32, contiguous k, same order for A and B): A lanes 0-15 K0..15,
    // lanes 16-31 K16..31; B [N][K] lanes 0-15 K0..15, 16-31 K16..31.
    const int kOffA = (lane < 16) ? 0 : 16;
    const int kOffB = (lane < 16) ? 0 : 16;
    const int l15   = lane & 15;

    bf16x16 bfrag[2];
#pragma unroll
    for (int j = 0; j < 2; ++j) {
      int n = wn * 32 + j * 16 + l15;
      bf16x8 lo = *(const bf16x8*)(Bs + n * LDS_K + kOffB);
      bf16x8 hi = *(const bf16x8*)(Bs + n * LDS_K + kOffB + 8);
      bfrag[j] = __builtin_shufflevector(lo, hi, 0,1,2,3,4,5,6,7,
                                                 8,9,10,11,12,13,14,15);
    }
    // One live A fragment at a time keeps VGPR pressure below spill point.
#pragma unroll
    for (int i = 0; i < MI; ++i) {
      int row = wm * (MI * 16) + i * 16 + l15;
      bf16x8 lo = *(const bf16x8*)(As + row * LDS_K + kOffA);
      bf16x8 hi = *(const bf16x8*)(As + row * LDS_K + 8 + kOffA);
      bf16x16 afrag = __builtin_shufflevector(lo, hi, 0,1,2,3,4,5,6,7,
                                                      8,9,10,11,12,13,14,15);
#pragma unroll
      for (int j = 0; j < 2; ++j)
        acc[i][j] = __builtin_amdgcn_wmma_f32_16x16x32_bf16(
            false, afrag, false, bfrag[j], (short)0, acc[i][j],
            false, false);
    }
    __syncthreads();
  }
}

// C-tile (16x16 f32): VGPR r -> row r (lanes 0-15) / r+8 (lanes 16-31);
// column = lane & 15.

// ---------------------------------------------------------------------------
// 1) QKV projection (TN=128). D scattered into per-head (bn,s,c) layout,
//    q scaled by 1/sqrt(c)=0.125 after bias (as in reference).
// ---------------------------------------------------------------------------
__global__ void __launch_bounds__(256, 1)
k_gemm_qkv(const float* __restrict__ A,
           const float* __restrict__ W,
           const float* __restrict__ bias,
           float* __restrict__ qh,
           float* __restrict__ kh,
           float* __restrict__ vh)
{
  __shared__ __bf16 As[TM * LDS_K];
  __shared__ __bf16 Bs[128 * LDS_K];
  f32x8 acc[4][2];
  const int mBase = blockIdx.y * TM;
  const int nBase = blockIdx.x * 128;
  wmma_pipe<4>(A, 1024, 20736, mBase, W, 3072, 1, nBase, 1024, As, Bs, acc);

  const int lane = threadIdx.x & 31, wid = threadIdx.x >> 5;
  const int wm = wid >> 2, wn = wid & 3;
  const int cbase = lane & 15;
  const int rplus = (lane < 16) ? 0 : 8;
#pragma unroll
  for (int i = 0; i < 4; ++i)
#pragma unroll
    for (int j = 0; j < 2; ++j) {
      int col  = nBase + wn * 32 + j * 16 + cbase;      // 0..3071
      int part = col >> 10, rem = col & 1023;
      int head = rem >> 6,  cc  = rem & 63;
      float* dst  = (part == 0) ? qh : (part == 1) ? kh : vh;
      float scale = (part == 0) ? 0.125f : 1.0f;
      float bv = bias[col];
#pragma unroll
      for (int r = 0; r < 8; ++r) {
        int row = mBase + wm * 64 + i * 16 + r + rplus; // 0..20735 (exact)
        int b = row / 5184, si = row - b * 5184;
        dst[(((size_t)(b * 16 + head)) * 5184 + si) * 64 + cc] =
            (acc[i][j][r] + bv) * scale;
      }
    }
}

// ---------------------------------------------------------------------------
// 2) sat = q_all @ k_text^T, masked. Batched over bn (grid.z). TN=64.
// ---------------------------------------------------------------------------
__global__ void k_gemm_sat(const float* __restrict__ qh,
                           const float* __restrict__ kh,
                           const float* __restrict__ mask,
                           float* __restrict__ sat)
{
  __shared__ __bf16 As[TM * LDS_K];
  __shared__ __bf16 Bs[64 * LDS_K];
  f32x8 acc[2][2];
  const int bn = blockIdx.z, b = bn >> 4;
  const int mBase = blockIdx.y * TM;
  const float* A = qh + (size_t)bn * 5184 * 64;
  const float* B = kh + (size_t)bn * 5184 * 64;     // rows 0..63 = k_text
  wmma_pipe<2>(A, 64, 5184, mBase, B, 64, 0, 0, 64, As, Bs, acc);

  const int lane = threadIdx.x & 31, wid = threadIdx.x >> 5;
  const int wm = wid >> 1, wn = wid & 1;
  const int cbase = lane & 15;
  const int rplus = (lane < 16) ? 0 : 8;
#pragma unroll
  for (int i = 0; i < 2; ++i)
#pragma unroll
    for (int j = 0; j < 2; ++j) {
      int col = wn * 32 + j * 16 + cbase;             // 0..63
#pragma unroll
      for (int r = 0; r < 8; ++r) {
        int row = mBase + wm * 32 + i * 16 + r + rplus;
        if (row < 5184) {
          float m = mask[((size_t)b * 5184 + row) * 64 + col];
          sat[((size_t)bn * 5184 + row) * 64 + col] =
              acc[i][j][r] * m - 10000.0f * (1.0f - m);
        }
      }
    }
}

// ---------------------------------------------------------------------------
// 3) ctx_text = probs_a2t @ v_text (B = vh text rows as [K=j][N=cc]). TN=64.
// ---------------------------------------------------------------------------
__global__ void k_gemm_ctx(const float* __restrict__ probs,
                           const float* __restrict__ vh,
                           float* __restrict__ ctx)
{
  __shared__ __bf16 As[TM * LDS_K];
  __shared__ __bf16 Bs[64 * LDS_K];
  f32x8 acc[2][2];
  const int bn = blockIdx.z, b = bn >> 4, head = bn & 15;
  const int mBase = blockIdx.y * TM;
  const float* A = probs + (size_t)bn * 5184 * 64;
  const float* B = vh + (size_t)bn * 5184 * 64;      // rows 0..63 = v_text
  wmma_pipe<2>(A, 64, 5184, mBase, B, 64, 1, 0, 64, As, Bs, acc);

  const int lane = threadIdx.x & 31, wid = threadIdx.x >> 5;
  const int wm = wid >> 1, wn = wid & 1;
  const int cbase = lane & 15;
  const int rplus = (lane < 16) ? 0 : 8;
#pragma unroll
  for (int i = 0; i < 2; ++i)
#pragma unroll
    for (int j = 0; j < 2; ++j) {
      int col = wn * 32 + j * 16 + cbase;             // cc 0..63
#pragma unroll
      for (int r = 0; r < 8; ++r) {
        int row = mBase + wm * 32 + i * 16 + r + rplus;
        if (row < 5184)
          ctx[((size_t)b * 5184 + row) * 1024 + head * 64 + col] = acc[i][j][r];
      }
    }
}

// ---------------------------------------------------------------------------
// 4) out = ctx @ w_out + b_out (TN=128).
// ---------------------------------------------------------------------------
__global__ void __launch_bounds__(256, 1)
k_gemm_out(const float* __restrict__ ctx,
           const float* __restrict__ W,
           const float* __restrict__ bias,
           float* __restrict__ out)
{
  __shared__ __bf16 As[TM * LDS_K];
  __shared__ __bf16 Bs[128 * LDS_K];
  f32x8 acc[4][2];
  const int mBase = blockIdx.y * TM;
  const int nBase = blockIdx.x * 128;
  wmma_pipe<4>(ctx, 1024, 20736, mBase, W, 1024, 1, nBase, 1024, As, Bs, acc);

  const int lane = threadIdx.x & 31, wid = threadIdx.x >> 5;
  const int wm = wid >> 2, wn = wid & 3;
  const int cbase = lane & 15;
  const int rplus = (lane < 16) ? 0 : 8;
#pragma unroll
  for (int i = 0; i < 4; ++i)
#pragma unroll
    for (int j = 0; j < 2; ++j) {
      int col = nBase + wn * 32 + j * 16 + cbase;
      float bv = bias[col];
#pragma unroll
      for (int r = 0; r < 8; ++r) {
        int row = mBase + wm * 64 + i * 16 + r + rplus;
        out[(size_t)row * 1024 + col] = acc[i][j][r] + bv;
      }
    }
}

// ---------------------------------------------------------------------------
// f_similar: one thread per (pixel, offset); float4 dot over c=64.
// Out-of-window positions hit the zero padding -> score 0 (matches ref).
// ---------------------------------------------------------------------------
__global__ void k_sim(const float* __restrict__ qh,
                      const float* __restrict__ kh,
                      float* __restrict__ out,
                      int qRow0, int kRow0, int Hq, int Wq, int Hk, int Wk,
                      int kH, int kW, int nOff)
{
  const int bn = blockIdx.z;
  const int npix = Hq * Wq;
  int t = blockIdx.x * blockDim.x + threadIdx.x;
  if (t >= npix * nOff) return;
  int pix = t / nOff, o = t - pix * nOff;
  int di = o / kW - (kH >> 1), dj = o % kW - (kW >> 1);
  int qr = pix / Wq, qc = pix - qr * Wq;
  int kr = qr / (Hq / Hk) + di, kc = qc / (Wq / Wk) + dj;
  float s = 0.0f;
  if (kr >= 0 && kr < Hk && kc >= 0 && kc < Wk) {
    const float* q = qh + ((size_t)bn * 5184 + qRow0 + pix) * 64;
    const float* k = kh + ((size_t)bn * 5184 + kRow0 + kr * Wk + kc) * 64;
#pragma unroll
    for (int c = 0; c < 64; c += 4) {
      float4 a = *(const float4*)(q + c);
      float4 b = *(const float4*)(k + c);
      s += a.x * b.x + a.y * b.y + a.z * b.z + a.w * b.w;
    }
  }
  out[((size_t)bn * npix + pix) * nOff + o] = s;
}

// ---------------------------------------------------------------------------
// Row softmax over concatenated [sat(64) | window scores], in place.
// One wave32 per row; <=154 values -> <=5 per lane.
// ---------------------------------------------------------------------------
__global__ void k_softmax(float* __restrict__ sat,
                          float* __restrict__ s00,
                          float* __restrict__ s10,
                          float* __restrict__ s11)
{
  const int bn = blockIdx.z;
  const int r  = blockIdx.x;
  const int lane = threadIdx.x;

  float* base = sat + ((size_t)bn * 5184 + r) * 64;
  float* e1 = nullptr; int n1 = 0;
  float* e2 = nullptr; int n2 = 0;
  if (r >= 1088) {
    e1 = s10 + ((size_t)bn * 4096 + (r - 1088)) * 49; n1 = 49;
    e2 = s11 + ((size_t)bn * 4096 + (r - 1088)) * 41; n2 = 41;
  } else if (r >= 64) {
    e1 = s00 + ((size_t)bn * 1024 + (r - 64)) * 41;   n1 = 41;
  }
  const int nv = 64 + n1 + n2;

  float vals[8];
  int cnt = 0;
  float mx = -1e30f;
  for (int idx = lane; idx < nv; idx += 32) {
    float v;
    if (idx < 64)            v = base[idx];
    else if (idx < 64 + n1)  v = e1[idx - 64];
    else                     v = e2[idx - 64 - n1];
    vals[cnt++] = v;
    mx = fmaxf(mx, v);
  }
  for (int m = 16; m >= 1; m >>= 1) mx = fmaxf(mx, __shfl_xor(mx, m, 32));
  float sum = 0.0f;
  for (int i = 0; i < cnt; ++i) { vals[i] = __expf(vals[i] - mx); sum += vals[i]; }
  for (int m = 16; m >= 1; m >>= 1) sum += __shfl_xor(sum, m, 32);
  const float inv = 1.0f / sum;
  cnt = 0;
  for (int idx = lane; idx < nv; idx += 32) {
    float v = vals[cnt++] * inv;
    if (idx < 64)            base[idx] = v;
    else if (idx < 64 + n1)  e1[idx - 64] = v;
    else                     e2[idx - 64 - n1] = v;
  }
}

// ---------------------------------------------------------------------------
// f_weighting: one thread per (pixel, 4 channels); accumulates into ctx.
// Kernels run sequentially on the stream, so += across c10/c11 is ordered.
// ---------------------------------------------------------------------------
__global__ void k_weight(const float* __restrict__ p,
                         const float* __restrict__ vh,
                         float* __restrict__ ctx,
                         int outRow0, int vRow0, int Hq, int Wq, int Hk, int Wk,
                         int kH, int kW, int nOff)
{
  const int bn = blockIdx.z, b = bn >> 4, head = bn & 15;
  const int npix = Hq * Wq;
  int t = blockIdx.x * blockDim.x + threadIdx.x;
  if (t >= npix * 16) return;
  int pix = t >> 4;
  int c4  = (t & 15) << 2;
  int qr = pix / Wq, qc = pix - qr * Wq;
  const int rdH = Hq / Hk, rdW = Wq / Wk;
  const float* prow = p + ((size_t)bn * npix + pix) * nOff;

  float ax = 0.f, ay = 0.f, az = 0.f, aw = 0.f;
  for (int o = 0; o < nOff; ++o) {
    int di = o / kW - (kH >> 1), dj = o % kW - (kW >> 1);
    int kr = qr / rdH + di, kc = qc / rdW + dj;
    if (kr < 0 || kr >= Hk || kc < 0 || kc >= Wk) continue;
    float w = prow[o];
    float4 v = *(const float4*)(vh + ((size_t)bn * 5184 + vRow0 +
                                      kr * Wk + kc) * 64 + c4);
    ax += w * v.x; ay += w * v.y; az += w * v.z; aw += w * v.w;
  }
  float* dst = ctx + ((size_t)b * 5184 + outRow0 + pix) * 1024 + head * 64 + c4;
  float4 cur = *(float4*)dst;
  cur.x += ax; cur.y += ay; cur.z += az; cur.w += aw;
  *(float4*)dst = cur;
}

// ---------------------------------------------------------------------------
extern "C" void kernel_launch(void* const* d_in, const int* in_sizes, int n_in,
                              void* d_out, int out_size, void* d_ws, size_t ws_size,
                              hipStream_t stream) {
  const float* hidden = (const float*)d_in[0];
  const float* mask   = (const float*)d_in[1];
  const float* w_qkv  = (const float*)d_in[2];
  const float* b_qkv  = (const float*)d_in[3];
  const float* w_out  = (const float*)d_in[4];
  const float* b_out  = (const float*)d_in[5];
  float* out = (float*)d_out;

  float* ws = (float*)d_ws;
  const size_t QH = 64ull * 5184 * 64;          // 21,233,664 floats
  float* qh  = ws;
  float* kh  = qh  + QH;
  float* vh  = kh  + QH;
  float* sat = vh  + QH;
  float* s00 = sat + QH;                        // 64*1024*41
  float* s10 = s00 + 64ull * 1024 * 41;         // 64*4096*49
  float* s11 = s10 + 64ull * 4096 * 49;         // 64*4096*41
  float* ctx = s11 + 64ull * 4096 * 41;         // 20736*1024

  const dim3 blk(256);

  // 1) QKV projection, scatter to per-head layout (q scaled by 1/8).
  k_gemm_qkv<<<dim3(24, 162, 1), blk, 0, stream>>>(hidden, w_qkv, b_qkv,
                                                   qh, kh, vh);
  // 2) Text attention scores (masked).
  k_gemm_sat<<<dim3(1, 41, 64), blk, 0, stream>>>(qh, kh, mask, sat);
  // 3) Local-window scores: s00 (9x9 causal), s10 (7x7 full), s11 (9x9 causal)
  k_sim<<<dim3(164, 1, 64), blk, 0, stream>>>(qh, kh, s00,
      64, 64, 32, 32, 32, 32, 9, 9, 41);
  k_sim<<<dim3(784, 1, 64), blk, 0, stream>>>(qh, kh, s10,
      1088, 64, 64, 64, 32, 32, 7, 7, 49);
  k_sim<<<dim3(656, 1, 64), blk, 0, stream>>>(qh, kh, s11,
      1088, 1088, 64, 64, 64, 64, 9, 9, 41);
  // 4) Joint softmax over [text | window] per row, in place.
  k_softmax<<<dim3(5184, 1, 64), dim3(32), 0, stream>>>(sat, s00, s10, s11);
  // 5) Text context (initializes ctx for all rows).
  k_gemm_ctx<<<dim3(1, 41, 64), blk, 0, stream>>>(sat, vh, ctx);
  // 6) Window weighting, accumulated into ctx.
  k_weight<<<dim3(64, 1, 64), blk, 0, stream>>>(s00, vh, ctx,
      64, 64, 32, 32, 32, 32, 9, 9, 41);
  k_weight<<<dim3(256, 1, 64), blk, 0, stream>>>(s10, vh, ctx,
      1088, 64, 64, 64, 32, 32, 7, 7, 49);
  k_weight<<<dim3(256, 1, 64), blk, 0, stream>>>(s11, vh, ctx,
      1088, 1088, 64, 64, 64, 64, 9, 9, 41);
  // 7) Output projection.
  k_gemm_out<<<dim3(8, 162, 1), blk, 0, stream>>>(ctx, w_out, b_out, out);
}
